// ImFace_37881611550835
// MI455X (gfx1250) — matrix operand order repaired and checked by reading, not compiled, not run on hardware
//
#include <hip/hip_runtime.h>
#include <hip/hip_bf16.h>

typedef __attribute__((ext_vector_type(16))) _Float16 v16h;
typedef __attribute__((ext_vector_type(8)))  float    v8f;
typedef __attribute__((ext_vector_type(4)))  unsigned int v4u;
typedef __attribute__((ext_vector_type(8)))  int          v8i;
typedef __attribute__((ext_vector_type(4)))  int          v4i;

#define NPTS 65536
// mini-MLP dims: din 204,128,128,128,128 (pad 224,128,...)
// deform dout ...,408(416) | ref ...,476(480) | tmpl ...,68(80)

// ---------------------------------------------------------------------------
// WMMA fragment layouts (CDNA5 ISA 7.12.2, wave32)
//  A 16x32 f16 : lane<16 row=lane, half h -> K=h; lane>=16: row=lane-16, K=16+h
//  B 32x16 f16 : half h -> K = (lane&16) + h, N = lane&15   (swizzled storage)
//  C/D 16x16 f32: c[i] -> M = i + 8*(lane>=16), N = lane&15
// Swizzled B storage: frag index = (nt*NK + kk)*32 + lane, 16 halves per frag.
// ---------------------------------------------------------------------------

template<int DINPAD, bool AF32, int EPI, bool PREF>
__device__ __forceinline__ void wave_gemm(const void* actIn, int ldIn,
                                          const _Float16* __restrict__ wswz,
                                          const float* __restrict__ bias,
                                          int dout, int doutPad,
                                          void* actOut, int ldOut, int lane)
{
    constexpr int NK = DINPAD / 32;
    const int m  = lane & 15;
    const int hb = (lane & 16) ? 8 : 0;
    v16h Af[NK];
#pragma unroll
    for (int kk = 0; kk < NK; ++kk) {
        v16h a;
#pragma unroll
        for (int hh = 0; hh < 16; ++hh) {
            int k = kk * 32 + 2 * hb + hh;
            if (AF32) a[hh] = (_Float16)(((const float*)actIn)[m * ldIn + k]);
            else      a[hh] = ((const _Float16*)actIn)[m * ldIn + k];
        }
        Af[kk] = a;
    }
    const int nN   = doutPad >> 4;
    const int nloc = lane & 15;
    const int mh   = (lane & 16) ? 8 : 0;
    const v16h* bbase = (const v16h*)wswz;
    for (int nt = 0; nt < nN; ++nt) {
        if (PREF && nt + 1 < nN)
            __builtin_prefetch((const void*)(bbase + (size_t)((nt + 1) * NK) * 32 + lane), 0, 3);
        v8f c = {};
#pragma unroll
        for (int kk = 0; kk < NK; ++kk) {
            v16h Bf = bbase[(size_t)(nt * NK + kk) * 32 + lane];
            c = __builtin_amdgcn_wmma_f32_16x16x32_f16(false, Af[kk], false, Bf,
                                                       (short)0, c, false, false);
        }
        int n = nt * 16 + nloc;
        float bv = (n < dout) ? bias[n] : 0.f;
#pragma unroll
        for (int i = 0; i < 8; ++i) {
            float v = c[i] + bv;
            if (EPI == 0) v = __sinf(30.f * v);
            int mm = i + mh;
            if (EPI == 2) ((float*)actOut)[mm * ldOut + n] = v;
            else          ((_Float16*)actOut)[mm * ldOut + n] = (_Float16)v;
        }
    }
}

// ---------------------------------------------------------------------------
// Tensor Data Mover: DMA `bytes/4` of a swizzled weight block into LDS per wave.
// D# built per CDNA5 ISA §8.3/8.4: count=1, type=2(image), data_size=8B, 1-D tile.
// Falls back to a plain vector copy if the builtin is unavailable (host pass).
// ---------------------------------------------------------------------------
__device__ __forceinline__ void tdm_copy_chunk(const _Float16* gsrc, _Float16* ldst,
                                               int totalBytes, int wv, int lane)
{
    const int chunk = totalBytes >> 2;                       // per-wave quarter
    const char* g = (const char*)gsrc + (size_t)wv * chunk;
    char*       l = (char*)ldst + (size_t)wv * chunk;
#if __has_builtin(__builtin_amdgcn_tensor_load_to_lds)
    (void)lane;
    unsigned long long ga = (unsigned long long)(const void*)g;
    unsigned int la = (unsigned int)(unsigned long long)(void*)l; // LDS byte offset
    unsigned int units = (unsigned int)(chunk >> 3);             // 8-byte elements
    v4u g0;
    g0[0] = 1u;                                                  // count=1, user D#
    g0[1] = la;                                                  // lds_addr [63:32]
    g0[2] = (unsigned int)(ga & 0xFFFFFFFFu);                    // global_addr lo
    g0[3] = (unsigned int)((ga >> 32) & 0x01FFFFFFu) | (2u << 30); // addr hi | type=2
    v8i g1;
    g1[0] = (int)(3u << 16);                                     // data_size=8B, mask=0
    g1[1] = (int)((units & 0xFFFFu) << 16);                      // tensor_dim0 lo16
    g1[2] = (int)(((units >> 16) & 0xFFFFu) | (1u << 16));       // dim0 hi16 | tensor_dim1=1
    g1[3] = (int)((units & 0xFFFFu) << 16);                      // tile_dim0
    g1[4] = 0;                                                   // tile_dim1/2 unused
    g1[5] = (int)units;                                          // tensor_dim0_stride lo32
    g1[6] = 0;
    g1[7] = 0;
    v4i gz = {0, 0, 0, 0};
#if __clang_major__ >= 23
    v8i gz8 = {0, 0, 0, 0, 0, 0, 0, 0};
    __builtin_amdgcn_tensor_load_to_lds(g0, g1, gz, gz, gz8, 0);
#else
    __builtin_amdgcn_tensor_load_to_lds(g0, g1, gz, gz, 0);
#endif
#else
    const uint4* s = (const uint4*)g;
    uint4*       d = (uint4*)l;
    for (int i = lane; i < (chunk >> 4); i += 32) d[i] = s[i];
#endif
}

__device__ __forceinline__ void tdm_wait()
{
#if __has_builtin(__builtin_amdgcn_s_wait_tensorcnt)
    __builtin_amdgcn_s_wait_tensorcnt(0);
#endif
}

// LayerNorm (+ optional leaky-relu) over row[0..len), 2 lanes/point, slots 128..131.
__device__ __forceinline__ void ln_act(float* row, const float* g, const float* b,
                                       int len, int h, bool lrelu)
{
    const int half = len >> 1;
    const int j0 = h * half;
    float s = 0.f, ss = 0.f;
    for (int j = j0; j < j0 + half; ++j) { float x = row[j]; s += x; ss += x * x; }
    row[128 + h] = s; row[130 + h] = ss;
    __syncthreads();
    float S = row[128] + row[129];
    float Q = row[130] + row[131];
    float mean = S / (float)len;
    float var  = Q / (float)len - mean * mean;
    float rstd = rsqrtf(var + 1e-5f);
    for (int j = j0; j < j0 + half; ++j) {
        float x = (row[j] - mean) * rstd * g[j] + b[j];
        if (lrelu) x = (x >= 0.f) ? x : 0.01f * x;
        row[j] = x;
    }
    __syncthreads();
}

__device__ __forceinline__ void se3_warp(const float* x, const float* d, float* o)
{
    float w0 = d[0], w1 = d[1], w2 = d[2];
    float v0 = d[3], v1 = d[4], v2 = d[5];
    float th = sqrtf(w0 * w0 + w1 * w1 + w2 * w2);
    float inv = 1.f / (th + 1e-8f);
    float a0 = w0 * inv, a1 = w1 * inv, a2 = w2 * inv;
    float b0 = v0 * inv, b1 = v1 * inv, b2 = v2 * inv;
    float st = __sinf(th), ct = __cosf(th);
    float omc = 1.f - ct, tms = th - st;
    float wdx = a0 * x[0] + a1 * x[1] + a2 * x[2];
    float c0 = a1 * x[2] - a2 * x[1];
    float c1 = a2 * x[0] - a0 * x[2];
    float c2 = a0 * x[1] - a1 * x[0];
    float R0 = x[0] + st * c0 + omc * (a0 * wdx - x[0]);
    float R1 = x[1] + st * c1 + omc * (a1 * wdx - x[1]);
    float R2 = x[2] + st * c2 + omc * (a2 * wdx - x[2]);
    float wdv = a0 * b0 + a1 * b1 + a2 * b2;
    float e0 = a1 * b2 - a2 * b1;
    float e1 = a2 * b0 - a0 * b2;
    float e2 = a0 * b1 - a1 * b0;
    float p0 = th * b0 + omc * e0 + tms * (a0 * wdv - b0);
    float p1 = th * b1 + omc * e1 + tms * (a1 * wdv - b1);
    float p2 = th * b2 + omc * e2 + tms * (a2 * wdv - b2);
    o[0] = R0 + p0; o[1] = R1 + p1; o[2] = R2 + p2;
}

// ---------------------------------------------------------------------------
// Prep kernels
// ---------------------------------------------------------------------------
__global__ __launch_bounds__(256) void prep_codes_kernel(
    const int* __restrict__ exp_idx, const int* __restrict__ id_idx,
    const float* __restrict__ exp_table, const float* __restrict__ id_table,
    const float* __restrict__ lm_p, const float* __restrict__ idlm_p,
    float* __restrict__ exp_e, float* __restrict__ id_e,
    float* __restrict__ lm_ori, float* __restrict__ lm_id)
{
    __shared__ float inb[256], h1[256], h2[256];
    int t = threadIdx.x;
    int ei = exp_idx[0], ii = id_idx[0];
    float cv = (t < 128) ? exp_table[(size_t)ei * 128 + t]
                         : id_table[(size_t)ii * 128 + (t - 128)];
    inb[t] = cv;
    if (t < 128) exp_e[t] = cv; else id_e[t - 128] = cv;
    __syncthreads();
    { const float* W = lm_p;          const float* b = lm_p + 65536;
      float s = b[t]; for (int i = 0; i < 256; ++i) s += inb[i] * W[(size_t)i * 256 + t];
      h1[t] = (s >= 0.f) ? s : 0.01f * s; }
    __syncthreads();
    { const float* W = lm_p + 65792;  const float* b = lm_p + 131328;
      float s = b[t]; for (int i = 0; i < 256; ++i) s += h1[i] * W[(size_t)i * 256 + t];
      h2[t] = (s >= 0.f) ? s : 0.01f * s; }
    __syncthreads();
    if (t < 204) {
      const float* W = lm_p + 131584; const float* b = lm_p + 183808;
      float s = b[t]; for (int i = 0; i < 256; ++i) s += h2[i] * W[(size_t)i * 204 + t];
      lm_ori[t] = s; }
    __syncthreads();
    { const float* W = idlm_p;         const float* b = idlm_p + 32768;
      float s = b[t]; for (int i = 0; i < 128; ++i) s += inb[128 + i] * W[(size_t)i * 256 + t];
      h1[t] = (s >= 0.f) ? s : 0.01f * s; }
    __syncthreads();
    { const float* W = idlm_p + 33024; const float* b = idlm_p + 98560;
      float s = b[t]; for (int i = 0; i < 256; ++i) s += h1[i] * W[(size_t)i * 256 + t];
      h2[t] = (s >= 0.f) ? s : 0.01f * s; }
    __syncthreads();
    if (t < 204) {
      const float* W = idlm_p + 98816; const float* b = idlm_p + 151040;
      float s = b[t]; for (int i = 0; i < 256; ++i) s += h2[i] * W[(size_t)i * 204 + t];
      lm_id[t] = s; }
}

__global__ __launch_bounds__(256) void gen_hyper_w_kernel(
    const float* __restrict__ code,
    const float* __restrict__ hw, const float* __restrict__ hb,
    const float* __restrict__ ow, const float* __restrict__ ob,
    _Float16* __restrict__ dst, int din, int dout, int dinPad, int doutPad)
{
    __shared__ float hv[128];
    int t = threadIdx.x;
    if (t < 128) {
        float s = hb[t];
        for (int i = 0; i < 128; ++i) s += code[i] * hw[(size_t)i * 128 + t];
        hv[t] = fmaxf(s, 0.f);
    }
    __syncthreads();
    int nK = dinPad >> 5;
    size_t dd = (size_t)din * dout;
    size_t total = (size_t)dinPad * doutPad;
    for (size_t idx = (size_t)blockIdx.x * 256 + t; idx < total;
         idx += (size_t)gridDim.x * 256) {
        int k = (int)(idx / doutPad);
        int n = (int)(idx % doutPad);
        float v = 0.f;
        if (k < din && n < dout) {
            size_t f = (size_t)k * dout + n;
            v = ob[f];
            for (int i = 0; i < 128; ++i) v += hv[i] * ow[(size_t)i * dd + f];
        }
        int kl = k & 31;
        size_t off = (((size_t)(n >> 4) * nK + (k >> 5)) * 32 + ((kl & 16) + (n & 15))) * 16 + (kl & 15);
        dst[off] = (_Float16)v;
    }
}

__global__ __launch_bounds__(256) void gen_hyper_b_kernel(
    const float* __restrict__ code,
    const float* __restrict__ hw, const float* __restrict__ hb,
    const float* __restrict__ ow, const float* __restrict__ ob,
    float* __restrict__ dst, int dout)
{
    __shared__ float hv[128];
    int t = threadIdx.x;
    if (t < 128) {
        float s = hb[t];
        for (int i = 0; i < 128; ++i) s += code[i] * hw[(size_t)i * 128 + t];
        hv[t] = fmaxf(s, 0.f);
    }
    __syncthreads();
    for (int n = blockIdx.x * 256 + t; n < dout; n += gridDim.x * 256) {
        float v = ob[n];
        for (int i = 0; i < 128; ++i) v += hv[i] * ow[(size_t)i * dout + n];
        dst[n] = v;
    }
}

__global__ __launch_bounds__(256) void conv_direct_kernel(
    const float* __restrict__ W, _Float16* __restrict__ dst,
    int din, int dout, int dinPad, int doutPad)
{
    int t = threadIdx.x;
    int nK = dinPad >> 5;
    size_t total = (size_t)dinPad * doutPad;
    for (size_t idx = (size_t)blockIdx.x * 256 + t; idx < total;
         idx += (size_t)gridDim.x * 256) {
        int k = (int)(idx / doutPad);
        int n = (int)(idx % doutPad);
        float v = (k < din && n < dout) ? W[(size_t)k * dout + n] : 0.f;
        int kl = k & 31;
        size_t off = (((size_t)(n >> 4) * nK + (k >> 5)) * 32 + ((kl & 16) + (n & 15))) * 16 + (kl & 15);
        dst[off] = (_Float16)v;
    }
}

// ---------------------------------------------------------------------------
// Per-point stage kernel: 128 threads = 4 waves, each wave owns 16 points.
// Layers 0-3 weight blocks are TDM-DMA'd into LDS (shared by the 4 waves);
// the wide layer 4 and fusion weights stream from L2.
// ---------------------------------------------------------------------------
struct FusionP {
    const float* l1W; const float* l1b;
    const float* g1;  const float* b1;
    const _Float16* l2swz; const float* l2b; const float* g2; const float* b2;
    const _Float16* l3swz; const float* l3b; const float* g3; const float* b3;
};
struct StageArgs {
    const float* xyz; const float* lm;
    const _Float16* w0; const _Float16* w1; const _Float16* w2;
    const _Float16* w3; const _Float16* w4;
    const float* mb0; const float* mb1; const float* mb2;
    const float* mb3; const float* mb4;
    int dout4; int dout4Pad;
    FusionP f;
    const float* fix_in;
    float* out; float* out_fix;
};

template<int STAGE>
__global__ __launch_bounds__(128) void stage_kernel(StageArgs sa)
{
    constexpr int CH = (STAGE == 0) ? 6 : ((STAGE == 1) ? 7 : 1);
    __shared__ __align__(16) _Float16 actS[4][16][480];
    __shared__ __align__(16) float    fusS[4][16][132];
    __shared__ __align__(16) _Float16 WBs[28672];          // 57344 B weight stage
    const int t    = threadIdx.x;
    const int wv   = t >> 5;
    const int lane = t & 31;
    const int p    = lane & 15;
    const int h    = (lane >> 4) & 1;
    const int base = blockIdx.x * 64 + wv * 16;
    _Float16 (*act)[480] = actS[wv];
    float    (*fus)[132] = fusS[wv];

    // start DMA of layer-0 weights, overlap with kx tile build
    tdm_copy_chunk(sa.w0, WBs, 57344, wv, lane);
    for (int idx = lane; idx < 16 * 224; idx += 32) {
        int m = idx / 224, j = idx - m * 224;
        float v = 0.f;
        if (j < 204) v = sa.xyz[(size_t)(base + m) * 3 + (j % 3)] - sa.lm[j];
        act[m][j] = (_Float16)v;
    }
    tdm_wait();
    __syncthreads();
    wave_gemm<224, false, 0, false>(act, 480, WBs, sa.mb0, 128, 128, act, 480, lane);
    __syncthreads();
    tdm_copy_chunk(sa.w1, WBs, 32768, wv, lane);
    tdm_wait();
    __syncthreads();
    wave_gemm<128, false, 0, false>(act, 480, WBs, sa.mb1, 128, 128, act, 480, lane);
    __syncthreads();
    tdm_copy_chunk(sa.w2, WBs, 32768, wv, lane);
    tdm_wait();
    __syncthreads();
    wave_gemm<128, false, 0, false>(act, 480, WBs, sa.mb2, 128, 128, act, 480, lane);
    __syncthreads();
    tdm_copy_chunk(sa.w3, WBs, 32768, wv, lane);
    tdm_wait();
    __syncthreads();
    wave_gemm<128, false, 0, false>(act, 480, WBs, sa.mb3, 128, 128, act, 480, lane);
    __syncthreads();
    wave_gemm<128, false, 1, true>(act, 480, sa.w4, sa.mb4, sa.dout4, sa.dout4Pad,
                                   act, 480, lane);
    __syncthreads();

    // -------- fusion network (2 lanes per point) --------
    const int gp = base + p;
    float px = sa.xyz[(size_t)gp * 3 + 0];
    float py = sa.xyz[(size_t)gp * 3 + 1];
    float pz = sa.xyz[(size_t)gp * 3 + 2];
    float* row = fus[p];
    for (int j = h * 64; j < h * 64 + 64; ++j)
        row[j] = px * sa.f.l1W[j] + py * sa.f.l1W[128 + j] + pz * sa.f.l1W[256 + j] + sa.f.l1b[j];
    __syncthreads();
    ln_act(row, sa.f.g1, sa.f.b1, 128, h, true);
    wave_gemm<128, true, 2, true>(fus, 132, sa.f.l2swz, sa.f.l2b, 128, 128, fus, 132, lane);
    __syncthreads();
    ln_act(row, sa.f.g2, sa.f.b2, 128, h, true);
    wave_gemm<128, true, 2, true>(fus, 132, sa.f.l3swz, sa.f.l3b, 68, 80, fus, 132, lane);
    __syncthreads();
    ln_act(row, sa.f.g3, sa.f.b3, 68, h, false);

    // softmax over 68 (unnormalized exp + denom)
    float mx = -3.0e38f;
    for (int j = h * 34; j < h * 34 + 34; ++j) mx = fmaxf(mx, row[j]);
    row[128 + h] = mx;
    __syncthreads();
    float M = fmaxf(row[128], row[129]);
    float s = 0.f;
    for (int j = h * 34; j < h * 34 + 34; ++j) { float e = __expf(row[j] - M); row[j] = e; s += e; }
    row[130 + h] = s;
    __syncthreads();
    float invden = 1.f / (row[130] + row[131]);

    // weighted reduction over K=68 landmarks
#pragma unroll
    for (int c = 0; c < CH; ++c) {
        float pc = 0.f;
        for (int k = h * 34; k < h * 34 + 34; ++k)
            pc += row[k] * (float)act[p][k * CH + c];
        row[112 + 2 * c + h] = pc;
    }
    __syncthreads();
    if (h == 0) {
        float dd[CH];
#pragma unroll
        for (int c = 0; c < CH; ++c)
            dd[c] = (row[112 + 2 * c] + row[113 + 2 * c]) * invden;
        if (STAGE <= 1) {
            float xi[3] = {px, py, pz}, o[3];
            se3_warp(xi, dd, o);
            sa.out[(size_t)gp * 3 + 0] = o[0];
            sa.out[(size_t)gp * 3 + 1] = o[1];
            sa.out[(size_t)gp * 3 + 2] = o[2];
            if (STAGE == 1) sa.out_fix[gp] = dd[CH - 1];
        } else {
            sa.out[gp] = dd[0] + sa.fix_in[gp];
        }
    }
}

// ---------------------------------------------------------------------------
// Host launcher
// ---------------------------------------------------------------------------
extern "C" void kernel_launch(void* const* d_in, const int* in_sizes, int n_in,
                              void* d_out, int out_size, void* d_ws, size_t ws_size,
                              hipStream_t stream)
{
    (void)in_sizes; (void)n_in; (void)out_size; (void)ws_size;
    const float* coords    = (const float*)d_in[0];
    const int*   exp_idx   = (const int*)d_in[1];
    const int*   id_idx    = (const int*)d_in[2];
    const float* exp_table = (const float*)d_in[3];
    const float* id_table  = (const float*)d_in[4];
    const float* lm_p      = (const float*)d_in[5];
    const float* idlm_p    = (const float*)d_in[6];
    const float* tmpl_lm   = (const float*)d_in[7];
    const float* dh        = (const float*)d_in[8];
    const float* dfu       = (const float*)d_in[9];
    const float* rh        = (const float*)d_in[10];
    const float* rfu       = (const float*)d_in[11];
    const float* tm        = (const float*)d_in[12];
    const float* tfu       = (const float*)d_in[13];

    const int din_[5]   = {204, 128, 128, 128, 128};
    const int dinP_[5]  = {224, 128, 128, 128, 128};
    const int doutD[5]  = {128, 128, 128, 128, 408};
    const int doutDP[5] = {128, 128, 128, 128, 416};
    const int doutR[5]  = {128, 128, 128, 128, 476};
    const int doutRP[5] = {128, 128, 128, 128, 480};
    const int doutT[5]  = {128, 128, 128, 128, 68};
    const int doutTP[5] = {128, 128, 128, 128, 80};

    char* wsb = (char*)d_ws;
    size_t off = 0;
    auto alloc = [&](size_t bytes) -> char* {
        char* pp = wsb + off;
        off += (bytes + 255) & ~(size_t)255;
        return pp;
    };
    float* exp_e  = (float*)alloc(128 * 4);
    float* id_e   = (float*)alloc(128 * 4);
    float* lm_ori = (float*)alloc(208 * 4);
    float* lm_id  = (float*)alloc(208 * 4);
    float* deformed = (float*)alloc((size_t)NPTS * 3 * 4);
    float* refered  = (float*)alloc((size_t)NPTS * 3 * 4);
    float* fixv     = (float*)alloc((size_t)NPTS * 4);
    _Float16 *dW[5], *rW[5], *tW[5];
    float *dB[5], *rB[5];
    for (int i = 0; i < 5; ++i) dW[i] = (_Float16*)alloc((size_t)dinP_[i] * doutDP[i] * 2);
    for (int i = 0; i < 5; ++i) rW[i] = (_Float16*)alloc((size_t)dinP_[i] * doutRP[i] * 2);
    for (int i = 0; i < 5; ++i) tW[i] = (_Float16*)alloc((size_t)dinP_[i] * doutTP[i] * 2);
    for (int i = 0; i < 5; ++i) dB[i] = (float*)alloc((size_t)doutDP[i] * 4);
    for (int i = 0; i < 5; ++i) rB[i] = (float*)alloc((size_t)doutRP[i] * 4);
    _Float16 *fu2[3], *fu3[3];
    for (int i = 0; i < 3; ++i) {
        fu2[i] = (_Float16*)alloc(128 * 128 * 2);
        fu3[i] = (_Float16*)alloc(128 * 80 * 2);
    }

    prep_codes_kernel<<<1, 256, 0, stream>>>(exp_idx, id_idx, exp_table, id_table,
                                             lm_p, idlm_p, exp_e, id_e, lm_ori, lm_id);

    struct HL { const float *bhW, *bhb, *boW, *bob, *whW, *whb, *woW, *wob; };
    HL dHL[5], rHL[5];
    auto fill = [&](const float* basep, const int* douts, HL* o) {
        size_t c = 0;
        for (int i = 0; i < 5; ++i) {
            size_t di = (size_t)din_[i], dd2 = (size_t)douts[i];
            o[i].bhW = basep + c; c += 128 * 128;
            o[i].bhb = basep + c; c += 128;
            o[i].boW = basep + c; c += 128 * dd2;
            o[i].bob = basep + c; c += dd2;
            o[i].whW = basep + c; c += 128 * 128;
            o[i].whb = basep + c; c += 128;
            o[i].woW = basep + c; c += 128 * di * dd2;
            o[i].wob = basep + c; c += di * dd2;
        }
    };
    fill(dh, doutD, dHL);
    fill(rh, doutR, rHL);

    for (int i = 0; i < 5; ++i) {
        int blocks = (dinP_[i] * doutDP[i] + 255) / 256;
        gen_hyper_w_kernel<<<blocks, 256, 0, stream>>>(exp_e, dHL[i].whW, dHL[i].whb,
            dHL[i].woW, dHL[i].wob, dW[i], din_[i], doutD[i], dinP_[i], doutDP[i]);
        gen_hyper_b_kernel<<<2, 256, 0, stream>>>(exp_e, dHL[i].bhW, dHL[i].bhb,
            dHL[i].boW, dHL[i].bob, dB[i], doutD[i]);
        blocks = (dinP_[i] * doutRP[i] + 255) / 256;
        gen_hyper_w_kernel<<<blocks, 256, 0, stream>>>(id_e, rHL[i].whW, rHL[i].whb,
            rHL[i].woW, rHL[i].wob, rW[i], din_[i], doutR[i], dinP_[i], doutRP[i]);
        gen_hyper_b_kernel<<<2, 256, 0, stream>>>(id_e, rHL[i].bhW, rHL[i].bhb,
            rHL[i].boW, rHL[i].bob, rB[i], doutR[i]);
    }

    const float* tb[5];
    { size_t c = 0;
      for (int i = 0; i < 5; ++i) {
          const float* Wp = tm + c; c += (size_t)din_[i] * doutT[i];
          tb[i] = tm + c;           c += doutT[i];
          int blocks = (dinP_[i] * doutTP[i] + 255) / 256;
          conv_direct_kernel<<<blocks, 256, 0, stream>>>(Wp, tW[i], din_[i], doutT[i],
                                                         dinP_[i], doutTP[i]);
      }
    }
    const float* fusb[3] = {dfu, rfu, tfu};
    for (int i = 0; i < 3; ++i) {
        conv_direct_kernel<<<64, 256, 0, stream>>>(fusb[i] + 1160,  fu2[i], 128, 128, 128, 128);
        conv_direct_kernel<<<40, 256, 0, stream>>>(fusb[i] + 17672, fu3[i], 128, 68, 128, 80);
    }

    auto mkFus = [&](const float* fb, const _Float16* s2, const _Float16* s3) {
        FusionP f;
        f.b1 = fb + 0;   f.b2 = fb + 128; f.b3 = fb + 256;
        f.g1 = fb + 324; f.g2 = fb + 452; f.g3 = fb + 580;
        f.l1W = fb + 648;  f.l1b = fb + 1032;
        f.l2swz = s2; f.l2b = fb + 17544;
        f.l3swz = s3; f.l3b = fb + 26376;
        return f;
    };

    const int nblk = NPTS / 64;
    {   StageArgs a{};
        a.xyz = coords; a.lm = lm_ori;
        a.w0 = dW[0]; a.w1 = dW[1]; a.w2 = dW[2]; a.w3 = dW[3]; a.w4 = dW[4];
        a.mb0 = dB[0]; a.mb1 = dB[1]; a.mb2 = dB[2]; a.mb3 = dB[3]; a.mb4 = dB[4];
        a.dout4 = 408; a.dout4Pad = 416;
        a.f = mkFus(dfu, fu2[0], fu3[0]);
        a.fix_in = nullptr; a.out = deformed; a.out_fix = nullptr;
        stage_kernel<0><<<nblk, 128, 0, stream>>>(a);
    }
    {   StageArgs a{};
        a.xyz = deformed; a.lm = lm_id;
        a.w0 = rW[0]; a.w1 = rW[1]; a.w2 = rW[2]; a.w3 = rW[3]; a.w4 = rW[4];
        a.mb0 = rB[0]; a.mb1 = rB[1]; a.mb2 = rB[2]; a.mb3 = rB[3]; a.mb4 = rB[4];
        a.dout4 = 476; a.dout4Pad = 480;
        a.f = mkFus(rfu, fu2[1], fu3[1]);
        a.fix_in = nullptr; a.out = refered; a.out_fix = fixv;
        stage_kernel<1><<<nblk, 128, 0, stream>>>(a);
    }
    {   StageArgs a{};
        a.xyz = refered; a.lm = tmpl_lm;
        a.w0 = tW[0]; a.w1 = tW[1]; a.w2 = tW[2]; a.w3 = tW[3]; a.w4 = tW[4];
        a.mb0 = tb[0]; a.mb1 = tb[1]; a.mb2 = tb[2]; a.mb3 = tb[3]; a.mb4 = tb[4];
        a.dout4 = 68; a.dout4Pad = 80;
        a.f = mkFus(tfu, fu2[2], fu3[2]);
        a.fix_in = fixv; a.out = (float*)d_out; a.out_fix = nullptr;
        stage_kernel<2><<<nblk, 128, 0, stream>>>(a);
    }
}
